// AttentionHead_37580963840263
// MI455X (gfx1250) — matrix operand order repaired and checked by reading, not compiled, not run on hardware
//
#include <hip/hip_runtime.h>
#include <hip/hip_bf16.h>

#define D_MODEL 1024
#define SEQ     2048
#define NBATCH  8
#define DKV     64

typedef __bf16 v16bf __attribute__((ext_vector_type(16)));
typedef __bf16 v8bf  __attribute__((ext_vector_type(8)));
typedef __bf16 v4bf  __attribute__((ext_vector_type(4)));
typedef float  v8f   __attribute__((ext_vector_type(8)));
typedef int    v4i   __attribute__((ext_vector_type(4)));

// ---- CDNA5 async global->LDS path (guarded; falls back to plain copy) ----
#if __has_builtin(__builtin_amdgcn_global_load_async_to_lds_b128)
#define HAVE_ASYNC_LDS 1
#endif

__device__ __forceinline__ void async_ld_b128(void* lds, const void* g) {
#if defined(HAVE_ASYNC_LDS)
    __builtin_amdgcn_global_load_async_to_lds_b128(
        (__attribute__((address_space(1))) v4i*)g,
        (__attribute__((address_space(3))) v4i*)lds, 0, 0);
#else
    *(uint4*)lds = *(const uint4*)g;
#endif
}

__device__ __forceinline__ void wait_async0() {
#if defined(HAVE_ASYNC_LDS)
#if __has_builtin(__builtin_amdgcn_s_wait_asynccnt)
    __builtin_amdgcn_s_wait_asynccnt(0);
#else
    asm volatile("s_wait_asynccnt 0x0" ::: "memory");
#endif
#endif
}

// ---- vector-load helpers for WMMA operand layouts (16B/32B aligned) ----
__device__ __forceinline__ v16bf ld16_split(const __bf16* p0, const __bf16* p1) {
    union { v16bf v; v8bf h[2]; } u;
    u.h[0] = *(const v8bf*)p0;
    u.h[1] = *(const v8bf*)p1;
    return u.v;
}
__device__ __forceinline__ v16bf ld16(const __bf16* p) {
    return *(const v16bf*)p;
}

// ---------------------------------------------------------------------------
// Pass 0: f32 -> bf16 conversion (memory bound; done once, not per GEMM tile)
// ---------------------------------------------------------------------------
__global__ void cvt_bf16(const float* __restrict__ in, __bf16* __restrict__ outp, int n)
{
    const int i = (blockIdx.x * blockDim.x + threadIdx.x) * 4;
    if (i + 3 < n) {
        const float4 f = *(const float4*)(in + i);
        v4bf o = { (__bf16)f.x, (__bf16)f.y, (__bf16)f.z, (__bf16)f.w };
        *(v4bf*)(outp + i) = o;
    }
}

// ---------------------------------------------------------------------------
// Pass 1: QKV projection on bf16 inputs.  One wave = 16-row x 64-col tile of
// one of Q/K/V.  A/B lane patterns are contiguous -> pure b128 vector loads.
// ---------------------------------------------------------------------------
__global__ __launch_bounds__(32) void qkv_proj(
    const __bf16* __restrict__ xb,
    const __bf16* __restrict__ Wq,
    const __bf16* __restrict__ Wk,
    const __bf16* __restrict__ Wv,
    __bf16* __restrict__ Qo,
    __bf16* __restrict__ Ko,
    __bf16* __restrict__ Vo)
{
    const int rowTile = blockIdx.x;
    const int which   = blockIdx.y;          // 0=Q, 1=K, 2=V
    const __bf16* W = (which == 0) ? Wq : (which == 1) ? Wk : Wv;
    __bf16*   outp  = (which == 0) ? Qo : (which == 1) ? Ko : Vo;
    const float sc  = (which == 0) ? 0.125f : 1.0f;   // fold 1/sqrt(64) into Q

    const int lane = threadIdx.x;
    const int half = lane >> 4;
    const int lm   = lane & 15;

    const __bf16* xrow = xb + (size_t)(rowTile * 16 + lm) * D_MODEL;
    const __bf16* w0 = W + (size_t)(0 * 16 + lm) * D_MODEL;
    const __bf16* w1 = W + (size_t)(1 * 16 + lm) * D_MODEL;
    const __bf16* w2 = W + (size_t)(2 * 16 + lm) * D_MODEL;
    const __bf16* w3 = W + (size_t)(3 * 16 + lm) * D_MODEL;

    v8f acc0 = {}, acc1 = {}, acc2 = {}, acc3 = {};

    for (int d0 = 0; d0 < D_MODEL; d0 += 32) {
        // A: K = half*16 + [0..15], contiguous 32B per lane
        const v16bf a = ld16(xrow + d0 + half * 16);
        // B: K = half*16 + [0..15], contiguous 32B per lane
        const v16bf b0 = ld16(w0 + d0 + half * 16);
        const v16bf b1 = ld16(w1 + d0 + half * 16);
        const v16bf b2 = ld16(w2 + d0 + half * 16);
        const v16bf b3 = ld16(w3 + d0 + half * 16);
        acc0 = __builtin_amdgcn_wmma_f32_16x16x32_bf16(false, a, false, b0, (short)0, acc0, false, false);
        acc1 = __builtin_amdgcn_wmma_f32_16x16x32_bf16(false, a, false, b1, (short)0, acc1, false, false);
        acc2 = __builtin_amdgcn_wmma_f32_16x16x32_bf16(false, a, false, b2, (short)0, acc2, false, false);
        acc3 = __builtin_amdgcn_wmma_f32_16x16x32_bf16(false, a, false, b3, (short)0, acc3, false, false);
    }

#pragma unroll
    for (int t = 0; t < 4; ++t) {
        const v8f acc = (t == 0) ? acc0 : (t == 1) ? acc1 : (t == 2) ? acc2 : acc3;
#pragma unroll
        for (int r = 0; r < 8; ++r) {
            const int orow = rowTile * 16 + r + half * 8;
            outp[(size_t)orow * DKV + t * 16 + lm] = (__bf16)(acc[r] * sc);
        }
    }
}

// ---------------------------------------------------------------------------
// Pass 2: causal flash attention.  One wave per (batch, 16-query tile).
// V chunks staged to LDS via async b128 DMA overlapped with the QK^T WMMAs.
// ---------------------------------------------------------------------------
__global__ __launch_bounds__(32) void flash_attn(
    const __bf16* __restrict__ Q,
    const __bf16* __restrict__ K,
    const __bf16* __restrict__ V,
    float* __restrict__ out)
{
    __shared__ __bf16 lds_p[16 * 32];       // P re-layout tile (1 KB)
    __shared__ __bf16 lds_v[32 * DKV];      // V chunk (4 KB)

    const int q0   = blockIdx.x * 16;
    const int b    = blockIdx.y;
    const int lane = threadIdx.x;
    const int half = lane >> 4;
    const int lm   = lane & 15;

    const __bf16* Qb = Q + (size_t)b * SEQ * DKV;
    const __bf16* Kb = K + (size_t)b * SEQ * DKV;
    const __bf16* Vb = V + (size_t)b * SEQ * DKV;

    // Q tile (16 x 64) as two A tiles
    v16bf qa0, qa1;
    {
        const __bf16* qrow = Qb + (size_t)(q0 + lm) * DKV;
        qa0 = ld16(qrow + half * 16);
        qa1 = ld16(qrow + 32 + half * 16);
    }

    float mrow[8], lrow[8];
    v8f oacc0 = {}, oacc1 = {}, oacc2 = {}, oacc3 = {};
#pragma unroll
    for (int r = 0; r < 8; ++r) { mrow[r] = -1e30f; lrow[r] = 0.0f; }

    const int kmax = q0 + 16;                 // causal bound (exclusive)
    for (int kt = 0; kt < kmax; kt += 32) {
        // ---- kick off async DMA of the V chunk (4KB), overlaps with QK^T ----
        {
            const char* vsrc = (const char*)(Vb + (size_t)kt * DKV);
            char*       vdst = (char*)lds_v;
#pragma unroll
            for (int i = 0; i < 8; ++i)
                async_ld_b128(vdst + i * 512 + lane * 16, vsrc + i * 512 + lane * 16);
        }

        // ---- scores: two 16x16 tiles, K-dim 64 -> 2 WMMAs each ----
        v8f s0 = {}, s1 = {};
        {
            const __bf16* k0 = Kb + (size_t)(kt + lm) * DKV;
            const __bf16* k1 = Kb + (size_t)(kt + 16 + lm) * DKV;
            const v16bf b0a = ld16(k0 + half * 16);
            const v16bf b0b = ld16(k0 + 32 + half * 16);
            const v16bf b1a = ld16(k1 + half * 16);
            const v16bf b1b = ld16(k1 + 32 + half * 16);
            s0 = __builtin_amdgcn_wmma_f32_16x16x32_bf16(false, qa0, false, b0a, (short)0, s0, false, false);
            s0 = __builtin_amdgcn_wmma_f32_16x16x32_bf16(false, qa1, false, b0b, (short)0, s0, false, false);
            s1 = __builtin_amdgcn_wmma_f32_16x16x32_bf16(false, qa0, false, b1a, (short)0, s1, false, false);
            s1 = __builtin_amdgcn_wmma_f32_16x16x32_bf16(false, qa1, false, b1b, (short)0, s1, false, false);
        }

        // ---- causal mask + tile row-max (cross-lane within 16-lane halves) ----
        float tm[8];
#pragma unroll
        for (int r = 0; r < 8; ++r) {
            const int q = q0 + r + half * 8;
            if (kt + lm > q)      s0[r] = -1e30f;
            if (kt + 16 + lm > q) s1[r] = -1e30f;
            float v = fmaxf(s0[r], s1[r]);
            v = fmaxf(v, __shfl_xor(v, 1, 16));
            v = fmaxf(v, __shfl_xor(v, 2, 16));
            v = fmaxf(v, __shfl_xor(v, 4, 16));
            v = fmaxf(v, __shfl_xor(v, 8, 16));
            tm[r] = v;
        }

        // ---- online softmax update ----
#pragma unroll
        for (int r = 0; r < 8; ++r) {
            const float mnew  = fmaxf(mrow[r], tm[r]);
            const float alpha = __expf(mrow[r] - mnew);
            const float p0 = __expf(s0[r] - mnew);
            const float p1 = __expf(s1[r] - mnew);
            s0[r] = p0; s1[r] = p1;
            float rs = p0 + p1;
            rs += __shfl_xor(rs, 1, 16);
            rs += __shfl_xor(rs, 2, 16);
            rs += __shfl_xor(rs, 4, 16);
            rs += __shfl_xor(rs, 8, 16);
            lrow[r] = lrow[r] * alpha + rs;
            mrow[r] = mnew;
            oacc0[r] *= alpha; oacc1[r] *= alpha; oacc2[r] *= alpha; oacc3[r] *= alpha;
        }

        // ---- P: C-layout -> LDS -> A-layout (bf16) ----
#pragma unroll
        for (int r = 0; r < 8; ++r) {
            const int rowi = r + half * 8;
            lds_p[rowi * 32 + lm]      = (__bf16)s0[r];
            lds_p[rowi * 32 + 16 + lm] = (__bf16)s1[r];
        }
        __syncthreads();
        v16bf pa;
#pragma unroll
        for (int j = 0; j < 8; ++j) {
            const int kk = half * 16 + 2 * j;
            pa[2 * j]     = lds_p[lm * 32 + kk];
            pa[2 * j + 1] = lds_p[lm * 32 + kk + 1];
        }

        // ---- V chunk ready?  wait for DMA, then O += P * V ----
        wait_async0();
        __syncthreads();
#pragma unroll
        for (int t = 0; t < 4; ++t) {
            v16bf vb;
#pragma unroll
            for (int j = 0; j < 8; ++j) {
                const int rr = half * 16 + 2 * j;
                vb[2 * j]     = lds_v[rr * DKV + t * 16 + lm];
                vb[2 * j + 1] = lds_v[(rr + 1) * DKV + t * 16 + lm];
            }
            v8f* op = (t == 0) ? &oacc0 : (t == 1) ? &oacc1 : (t == 2) ? &oacc2 : &oacc3;
            *op = __builtin_amdgcn_wmma_f32_16x16x32_bf16(
                false, pa, false, vb, (short)0, *op, false, false);
        }
        __syncthreads();   // all lanes done with lds_v before next chunk's DMA
    }

    // ---- epilogue: O / l ----
    float* ob = out + ((size_t)b * SEQ + q0) * DKV;
#pragma unroll
    for (int r = 0; r < 8; ++r) {
        const float inv  = 1.0f / lrow[r];
        const int   rowi = r + half * 8;
        ob[(size_t)rowi * DKV + 0 * 16 + lm] = oacc0[r] * inv;
        ob[(size_t)rowi * DKV + 1 * 16 + lm] = oacc1[r] * inv;
        ob[(size_t)rowi * DKV + 2 * 16 + lm] = oacc2[r] * inv;
        ob[(size_t)rowi * DKV + 3 * 16 + lm] = oacc3[r] * inv;
    }
}

// ---------------------------------------------------------------------------
extern "C" void kernel_launch(void* const* d_in, const int* in_sizes, int n_in,
                              void* d_out, int out_size, void* d_ws, size_t ws_size,
                              hipStream_t stream)
{
    (void)in_sizes; (void)n_in; (void)out_size; (void)ws_size;
    const float* x  = (const float*)d_in[0];
    const float* Wq = (const float*)d_in[1];
    const float* Wk = (const float*)d_in[2];
    const float* Wv = (const float*)d_in[3];

    const size_t NX = (size_t)NBATCH * SEQ * D_MODEL;   // 16,777,216
    const size_t NW = (size_t)DKV * D_MODEL;            // 65,536
    const size_t NQ = (size_t)NBATCH * SEQ * DKV;       // 1,048,576

    __bf16* xbf = (__bf16*)d_ws;
    __bf16* wqb = xbf + NX;
    __bf16* wkb = wqb + NW;
    __bf16* wvb = wkb + NW;
    __bf16* Qo  = wvb + NW;
    __bf16* Ko  = Qo + NQ;
    __bf16* Vo  = Ko + NQ;

    // Pass 0: one-time f32 -> bf16 conversion
    cvt_bf16<<<dim3((unsigned)(NX / 4 / 256)), 256, 0, stream>>>(x, xbf, (int)NX);
    cvt_bf16<<<dim3((unsigned)(NW / 4 / 64)), 64, 0, stream>>>(Wq, wqb, (int)NW);
    cvt_bf16<<<dim3((unsigned)(NW / 4 / 64)), 64, 0, stream>>>(Wk, wkb, (int)NW);
    cvt_bf16<<<dim3((unsigned)(NW / 4 / 64)), 64, 0, stream>>>(Wv, wvb, (int)NW);

    // Pass 1: QKV projection (3072 waves)
    dim3 g1(NBATCH * SEQ / 16, 3);
    qkv_proj<<<g1, 32, 0, stream>>>(xbf, wqb, wkb, wvb, Qo, Ko, Vo);

    // Pass 2: causal flash attention (1024 waves)
    dim3 g2(SEQ / 16, NBATCH);
    flash_attn<<<g2, 32, 0, stream>>>(Qo, Ko, Vo, (float*)d_out);
}
